// GraphEncoder_1520418422912
// MI455X (gfx1250) — matrix operand order repaired and checked by reading, not compiled, not run on hardware
//
#include <hip/hip_runtime.h>

typedef __attribute__((ext_vector_type(16))) _Float16 v16h;
typedef __attribute__((ext_vector_type(2)))  _Float16 h2;
typedef __attribute__((ext_vector_type(8)))  float    v8f;

static constexpr int kN   = 50000;
static constexpr int kE   = 800000;
static constexpr int kIN  = 8;
static constexpr int kH   = 16;
static constexpr int kOUT = 16;
static constexpr int kB   = 64;

// ---- workspace layout (bytes) ----
static constexpr size_t FRAG_OFF = 0;                       // 14 chunks * 32 lanes * 16 halves * 2B
static constexpr size_t FRAG_BYTES = 14ull * 32 * 16 * 2;   // 14336
static constexpr size_t AGG1_OFF = FRAG_OFF + FRAG_BYTES;   // N*16 f32
static constexpr size_t CNT1_OFF = AGG1_OFF + (size_t)kN * 16 * 4;
static constexpr size_t H1_OFF   = CNT1_OFF + (size_t)kN * 4;
static constexpr size_t AGG2_OFF = H1_OFF   + (size_t)kN * 16 * 4;
static constexpr size_t CNT2_OFF = AGG2_OFF + (size_t)kN * 16 * 4;
static constexpr size_t GSUM_OFF = CNT2_OFF + (size_t)kN * 4;
static constexpr size_t GCNT_OFF = GSUM_OFF + (size_t)kB * 16 * 4;
static constexpr size_t WS_END   = GCNT_OFF + (size_t)kB * 4;
static constexpr int    ZERO_FLOATS = (int)((WS_END - AGG1_OFF) / 4);

__device__ __forceinline__ void atomAdd(float* p, float v) {
  unsafeAtomicAdd(p, v);   // hardware global_atomic_add_f32
}

// --------------------------------------------------------------------------
// Precompute shared B-matrix fragments (WMMA B layout) for both layers.
// Mext1[k][o], k=i*16+j (k<128) -> W1b[(i*16+o)*16+j]; k in [128,136) -> b1b[(k-128)*16+o]
// Mext2[k][o], k=i*16+j (k<256) -> W2b[(i*16+o)*16+j]; k in [256,272) -> b2b[(k-256)*16+o]
// frag[((chunk*32)+lane)*16 + slot] = Mext[32*chunk + (lane<16?0:8) + slot + (slot&8)][lane&15]
// Chunks 0..4 = layer1, 5..13 = layer2.
// --------------------------------------------------------------------------
__global__ void prep_frags(const float* __restrict__ W1b, const float* __restrict__ b1b,
                           const float* __restrict__ W2b, const float* __restrict__ b2b,
                           _Float16* __restrict__ frag) {
  const int total = 14 * 32 * 16;
  for (int t = blockIdx.x * blockDim.x + threadIdx.x; t < total;
       t += gridDim.x * blockDim.x) {
    const int s = t & 15;
    const int L = (t >> 4) & 31;
    const int c = t >> 9;
    const int o = L & 15;
    const int kb = (L < 16) ? 0 : 8;
    float val = 0.f;
    if (c < 5) {
      const int k = 32 * c + kb + s + (s & 8);
      if (k < 128)       val = W1b[((k >> 4) * 16 + o) * 16 + (k & 15)];
      else if (k < 136)  val = b1b[(k - 128) * 16 + o];
    } else {
      const int k = 32 * (c - 5) + kb + s + (s & 8);
      if (k < 256)       val = W2b[((k >> 4) * 16 + o) * 16 + (k & 15)];
      else if (k < 272)  val = b2b[(k - 256) * 16 + o];
    }
    frag[t] = (_Float16)val;
  }
}

__global__ void zero_f32(float* __restrict__ p, int n) {
  int t = blockIdx.x * blockDim.x + threadIdx.x;
  const int stride = gridDim.x * blockDim.x;
  for (; t < n; t += stride) p[t] = 0.f;
}

// --------------------------------------------------------------------------
// Edge pass: each wave handles 16 edges.  msg = Z(16xK) @ Mext(Kx16) with
// z = outer(x_src, he) built directly in f16 via v_pk_mul_f16, consumed by
// chained v_wmma_f32_16x16x32_f16; results scatter-added to agg[dst].
// --------------------------------------------------------------------------
template <int DIN>
__global__ void edge_pass(const float* __restrict__ feat,   // [N, DIN] source features
                          const float* __restrict__ ea,     // [E] edge attr
                          const float* __restrict__ wa,     // [16] edge-MLP lin1 weight
                          const float* __restrict__ ba,     // [16] edge-MLP lin1 bias
                          const int*   __restrict__ ei,     // [2, E] edge index
                          const _Float16* __restrict__ frag,// B fragments for this layer
                          float* __restrict__ agg,          // [N,16] accumulators
                          float* __restrict__ cnt) {        // [N] edge counts
  constexpr int KZ = DIN * 16;        // outer-product width
  constexpr int KB = KZ + DIN;        // + bias-carrier rows
  constexpr int CH = (KB + 31) / 32;  // K chunks of 32

  const int lane   = threadIdx.x & 31;
  const int wave   = (blockIdx.x * blockDim.x + threadIdx.x) >> 5;
  const int ebase  = wave << 4;
  const int m      = lane & 15;
  const int e      = ebase + m;
  const int laneHi = lane >> 4;

  const int   srcN = ei[e];
  const float a    = ea[e];

  // Source features, converted once to f16 (scalar + pair forms).
  _Float16 fs16[DIN];
#pragma unroll
  for (int i = 0; i < DIN / 4; ++i) {
    float4 v4 = ((const float4*)(feat + srcN * DIN))[i];
    fs16[4 * i + 0] = (_Float16)v4.x; fs16[4 * i + 1] = (_Float16)v4.y;
    fs16[4 * i + 2] = (_Float16)v4.z; fs16[4 * i + 3] = (_Float16)v4.w;
  }
  h2 fsp[DIN / 2];
#pragma unroll
  for (int i = 0; i < DIN / 2; ++i) fsp[i] = h2{fs16[2 * i], fs16[2 * i + 1]};

  // he = relu(a*wa + ba) as 8 packed f16 pairs.
  const _Float16 a16 = (_Float16)a;
  const _Float16 z16 = (_Float16)0;
  h2 hep[8];
#pragma unroll
  for (int t = 0; t < 8; ++t) {
    h2 w = h2{(_Float16)wa[2 * t], (_Float16)wa[2 * t + 1]};
    h2 b = h2{(_Float16)ba[2 * t], (_Float16)ba[2 * t + 1]};
    h2 v = h2{a16, a16} * w + b;           // v_pk_fma_f16
    v.x = v.x > z16 ? v.x : z16;
    v.y = v.y > z16 ? v.y : z16;
    hep[t] = v;
  }

  v8f acc = {};
  const v16h* bfrags = (const v16h*)frag;
  v16h afbuf[2];                            // double-buffer A frags (avoid WMMA WAR nops)
#pragma unroll
  for (int c = 0; c < CH; ++c) {
    v16h& af = afbuf[c & 1];
#pragma unroll
    for (int v = 0; v < 8; ++v) {
      // 16-bit A-matrix 16x32 layout; k0 is the (even) first K of this half2 slot
      const int kloc0 = (v < 4) ? (laneHi * 8 + 2 * v)
                                : (16 + laneHi * 8 + 2 * (v - 4));
      const int k0 = 32 * c + kloc0;
      h2 p = h2{z16, z16};
      if (k0 < KZ) {
        const int i  = (k0 >> 4) & (DIN - 1);
        const int jp = (k0 & 15) >> 1;
        p = h2{fs16[i], fs16[i]} * hep[jp]; // v_pk_mul_f16
      } else if (k0 < KB) {
        p = fsp[((k0 - KZ) >> 1) & (DIN / 2 - 1)];
      }
      af[2 * v]     = p.x;
      af[2 * v + 1] = p.y;
    }
    const v16h bf = bfrags[c * 32 + lane];
    acc = __builtin_amdgcn_wmma_f32_16x16x32_f16(false, af, false, bf,
                                                 (short)0, acc, false, false);
  }

  // C/D layout: element r -> edge row (laneHi?8+r:r), column o = lane&15
  const int o = lane & 15;
#pragma unroll
  for (int r = 0; r < 8; ++r) {
    const int mo = laneHi * 8 + r;
    const int dn = ei[kE + ebase + mo];
    atomAdd(&agg[dn * 16 + o], acc[r]);
  }
  if (lane < 16) {
    atomAdd(&cnt[ei[kE + e]], 1.0f);
  }
}

// h[n,o] = relu(agg/max(cnt,1) + feat @ root + bias)
template <int DIN>
__global__ void node_pass(const float* __restrict__ feat,
                          const float* __restrict__ agg,
                          const float* __restrict__ cnt,
                          const float* __restrict__ root,
                          const float* __restrict__ bias,
                          float* __restrict__ outp) {
  const int t = blockIdx.x * blockDim.x + threadIdx.x;
  if (t >= kN * 16) return;
  const int n = t >> 4, o = t & 15;
  float c = cnt[n]; c = c > 1.f ? c : 1.f;
  float v = agg[t] / c + bias[o];
#pragma unroll
  for (int i = 0; i < DIN; ++i)
    v = fmaf(feat[n * DIN + i], root[i * 16 + o], v);
  outp[t] = v > 0.f ? v : 0.f;
}

__global__ void pool_scatter(const float* __restrict__ h2v,
                             const int* __restrict__ batch,
                             float* __restrict__ gsum, float* __restrict__ gcnt,
                             float* __restrict__ outBatch) {
  const int t = blockIdx.x * blockDim.x + threadIdx.x;
  if (t >= kN * 16) return;
  const int n = t >> 4, o = t & 15;
  const int b = batch[n];
  atomAdd(&gsum[b * 16 + o], h2v[t]);
  if (o == 0) {
    atomAdd(&gcnt[b], 1.0f);
    outBatch[n] = (float)b;
  }
}

__global__ void pool_final(const float* __restrict__ gsum,
                           const float* __restrict__ gcnt,
                           float* __restrict__ gout) {
  const int t = threadIdx.x;
  if (t < kB * 16) {
    float c = gcnt[t >> 4]; c = c > 1.f ? c : 1.f;
    gout[t] = gsum[t] / c;
  }
}

extern "C" void kernel_launch(void* const* d_in, const int* in_sizes, int n_in,
                              void* d_out, int out_size, void* d_ws, size_t ws_size,
                              hipStream_t stream) {
  const float* x     = (const float*)d_in[0];
  const float* ea    = (const float*)d_in[1];
  const float* w1a   = (const float*)d_in[2];
  const float* b1a   = (const float*)d_in[3];
  const float* W1b   = (const float*)d_in[4];
  const float* b1b   = (const float*)d_in[5];
  const float* root1 = (const float*)d_in[6];
  const float* bias1 = (const float*)d_in[7];
  const float* w2a   = (const float*)d_in[8];
  const float* b2a   = (const float*)d_in[9];
  const float* W2b   = (const float*)d_in[10];
  const float* b2b   = (const float*)d_in[11];
  const float* root2 = (const float*)d_in[12];
  const float* bias2 = (const float*)d_in[13];
  const int*   ei    = (const int*)d_in[14];
  const int*   batch = (const int*)d_in[15];
  float* out = (float*)d_out;

  char* ws = (char*)d_ws;
  _Float16* frag = (_Float16*)(ws + FRAG_OFF);
  float* agg1 = (float*)(ws + AGG1_OFF);
  float* cnt1 = (float*)(ws + CNT1_OFF);
  float* h1   = (float*)(ws + H1_OFF);
  float* agg2 = (float*)(ws + AGG2_OFF);
  float* cnt2 = (float*)(ws + CNT2_OFF);
  float* gsum = (float*)(ws + GSUM_OFF);
  float* gcnt = (float*)(ws + GCNT_OFF);

  prep_frags<<<28, 256, 0, stream>>>(W1b, b1b, W2b, b2b, frag);
  zero_f32<<<2048, 256, 0, stream>>>(agg1, ZERO_FLOATS);

  // Layer 1: 16 edges per wave, 5 chained wmma (K = 136 padded to 160)
  edge_pass<kIN><<<kE / 128, 256, 0, stream>>>(x, ea, w1a, b1a, ei, frag, agg1, cnt1);
  node_pass<kIN><<<(kN * 16 + 255) / 256, 256, 0, stream>>>(x, agg1, cnt1, root1, bias1, h1);

  // Layer 2: 9 chained wmma (K = 272 padded to 288); h2 written straight to d_out
  edge_pass<kH><<<kE / 128, 256, 0, stream>>>(h1, ea, w2a, b2a, ei,
                                              frag + 5 * 32 * 16, agg2, cnt2);
  node_pass<kH><<<(kN * 16 + 255) / 256, 256, 0, stream>>>(h1, agg2, cnt2, root2, bias2, out);

  // Global mean pool + batch passthrough
  pool_scatter<<<(kN * 16 + 255) / 256, 256, 0, stream>>>(out, batch, gsum, gcnt,
                                                          out + (size_t)kN * 16 + kB * 16);
  pool_final<<<1, kB * 16, 0, stream>>>(gsum, gcnt, out + (size_t)kN * 16);
}
